// MoELayerOptimized_14860586844371
// MI455X (gfx1250) — matrix operand order repaired and checked by reading, not compiled, not run on hardware
//
#include <hip/hip_runtime.h>
#include <cstddef>
#include <cstdint>

// ---------------------------------------------------------------------------
// MoE layer for MI455X (gfx1250, wave32, WMMA, double-buffered async-to-LDS).
//   T=4096 tokens, D=1024, E=8 experts (+1 shared), I=938 (padded to 960), K=2.
// Pipeline:
//   1) router_convert : logits/softmax/top2/renorm -> combine[t][9], x -> f16
//   2) transpose_gu   : gate/up weights -> f16, transposed [n][d], zero-padded
//   3) transpose_dn   : down weights    -> f16, transposed [n][9*960]
//   4) gemm1          : h'[t][e*960+i] = combine[t][e]*silu(g)*u   (WMMA f16)
//   5) gemm2          : out = h' @ Wdn_t^T  (single GEMM, K = 9*960 = 8640)
// Steady-state K-loop is unrolled x2 so LDS buffer selection is static
// (no cndmask / no branch); only the peeled tail waits asynccnt to 0.
// ---------------------------------------------------------------------------

typedef __attribute__((ext_vector_type(16))) _Float16 v16h;
typedef __attribute__((ext_vector_type(8)))  float    v8f;

#define T_TOK   4096
#define DDIM    1024
#define IACT    938
#define IPAD    960
#define NEXP    9            // shared + 8 routed
#define KTOT    (NEXP*IPAD)  // 8640
#define CSTRIDE 16
#define LDK     40           // LDS row pitch in halves (80B, conflict-free, 16B aligned)

// ---------------- CDNA5 async global->LDS copy (ASYNCcnt-tracked) ----------
// VDST = workgroup-relative LDS byte offset (= low 32 bits of flat LDS addr).
static __device__ __forceinline__ void async_ld16(uint32_t lds_byte, const _Float16* g) {
  asm volatile("global_load_async_to_lds_b128 %0, %1, off"
               :: "v"(lds_byte), "v"(g) : "memory");
}
// Wait until <=4 async ops outstanding: the 4 newest (next tile) may still fly,
// the previous tile's 4 are guaranteed complete (async loads complete in order).
static __device__ __forceinline__ void async_wait4() {
  asm volatile("s_wait_asynccnt 0x4" ::: "memory");
}
static __device__ __forceinline__ void async_wait0() {
  asm volatile("s_wait_asynccnt 0x0" ::: "memory");
}
static __device__ __forceinline__ uint32_t lds_off(const void* p) {
  return (uint32_t)(uintptr_t)p;
}

// --------------------------- WMMA fragment loads ---------------------------
// A (16x32 f16): lane<16 -> M=lane, halves K0..7 | K16..23 ; lane>=16 -> K8..15 | K24..31
static __device__ __forceinline__ v16h frag_a(const _Float16* base, int lane) {
  const int row = lane & 15;
  const _Float16* p = base + row * LDK + ((lane & 16) ? 8 : 0);
  union { float4 f[2]; v16h v; } u;
  u.f[0] = *(const float4*)(p);
  u.f[1] = *(const float4*)(p + 16);
  return u.v;
}
// B (32x16 f16), stored [n][k]: lane<16 -> N=lane, K0..7 | K16..23 ; lane>=16 -> N=lane-16, K8..15 | K24..31
static __device__ __forceinline__ v16h frag_b(const _Float16* base, int lane) {
  const int col = lane & 15;
  const _Float16* p = base + col * LDK + ((lane & 16) ? 8 : 0);
  union { float4 f[2]; v16h v; } u;
  u.f[0] = *(const float4*)(p);
  u.f[1] = *(const float4*)(p + 16);
  return u.v;
}

// --------------------------- 1) router + x->f16 ----------------------------
__global__ __launch_bounds__(128) void router_convert(
    const float* __restrict__ x, const float* __restrict__ gw,
    _Float16* __restrict__ xf16, float* __restrict__ comb) {
  const int t   = blockIdx.x;
  const int tid = threadIdx.x;
  __shared__ float part[8][129];
  float acc[8];
#pragma unroll
  for (int e = 0; e < 8; ++e) acc[e] = 0.f;
  const float* xr = x + (size_t)t * DDIM;
  for (int d = tid; d < DDIM; d += 128) {
    const float xv = xr[d];
    xf16[(size_t)t * DDIM + d] = (_Float16)xv;
#pragma unroll
    for (int e = 0; e < 8; ++e) acc[e] += xv * gw[e * DDIM + d];
  }
#pragma unroll
  for (int e = 0; e < 8; ++e) part[e][tid] = acc[e];
  __syncthreads();
  if (tid < 8) {
    float s = 0.f;
    for (int j = 0; j < 128; ++j) s += part[tid][j];
    part[tid][128] = s;
  }
  __syncthreads();
  if (tid == 0) {
    float l[8], m = -1e30f;
#pragma unroll
    for (int e = 0; e < 8; ++e) { l[e] = part[e][128]; m = fmaxf(m, l[e]); }
    float p[8], sum = 0.f;
#pragma unroll
    for (int e = 0; e < 8; ++e) { p[e] = __expf(l[e] - m); sum += p[e]; }
    const float inv = 1.0f / sum;
#pragma unroll
    for (int e = 0; e < 8; ++e) p[e] *= inv;
    int i0 = 0;
#pragma unroll
    for (int e = 1; e < 8; ++e) if (p[e] > p[i0]) i0 = e;
    int i1 = (i0 == 0) ? 1 : 0;
#pragma unroll
    for (int e = 0; e < 8; ++e) if (e != i0 && p[e] > p[i1]) i1 = e;
    const float denom = p[i0] + p[i1] + 1e-8f;
    float* c = comb + (size_t)t * CSTRIDE;
    c[0] = 1.0f;                       // shared expert, always on
#pragma unroll
    for (int e = 0; e < 8; ++e) c[1 + e] = 0.f;
    c[1 + i0] = p[i0] / denom;
    c[1 + i1] = p[i1] / denom;
  }
}

// ------------------- 2) gate/up weight transpose (f32->f16) ----------------
// dst: wgut[e][n][d], n in [0,1920): n<960 -> gate col n ; n>=960 -> up col n-960.
// src col = h*938 + i ; zero-fill i in [938,960).
__global__ __launch_bounds__(256) void transpose_gu(
    const float* __restrict__ sgu, const float* __restrict__ egu,
    _Float16* __restrict__ wgut) {
  __shared__ float tile[32][33];
  const int e = blockIdx.z;
  const float* src = (e == 0) ? sgu : egu + (size_t)(e - 1) * DDIM * (2 * IACT);
  const int nBase = blockIdx.x * 32;      // 0..1919
  const int dBase = blockIdx.y * 32;
  const int tx = threadIdx.x, ty = threadIdx.y;
  const int h  = nBase / IPAD;            // tiles never straddle the 960 boundary
  const int i  = (nBase - h * IPAD) + tx;
  const bool valid = (i < IACT);
  const int scol = h * IACT + i;
#pragma unroll
  for (int j = 0; j < 4; ++j) {
    const int d = dBase + ty + j * 8;
    tile[ty + j * 8][tx] = valid ? src[(size_t)d * (2 * IACT) + scol] : 0.0f;
  }
  __syncthreads();
#pragma unroll
  for (int j = 0; j < 4; ++j) {
    const int nl = ty + j * 8;
    wgut[((size_t)e * 1920 + nBase + nl) * DDIM + dBase + tx] = (_Float16)tile[tx][nl];
  }
}

// -------------------- 3) down weight transpose (f32->f16) ------------------
// dst: wdnt[n][e*960 + i] ; zero-fill i in [938,960).
__global__ __launch_bounds__(256) void transpose_dn(
    const float* __restrict__ sdn, const float* __restrict__ edn,
    _Float16* __restrict__ wdnt) {
  __shared__ float tile[32][33];
  const int e = blockIdx.z;
  const float* src = (e == 0) ? sdn : edn + (size_t)(e - 1) * IACT * DDIM;
  const int iBase = blockIdx.x * 32;      // 0..959
  const int nBase = blockIdx.y * 32;
  const int tx = threadIdx.x, ty = threadIdx.y;
#pragma unroll
  for (int j = 0; j < 4; ++j) {
    const int irow = iBase + ty + j * 8;
    tile[ty + j * 8][tx] = (irow < IACT) ? src[(size_t)irow * DDIM + nBase + tx] : 0.0f;
  }
  __syncthreads();
#pragma unroll
  for (int j = 0; j < 4; ++j) {
    const int nl = ty + j * 8;
    wdnt[(size_t)(nBase + nl) * KTOT + e * IPAD + iBase + tx] = (_Float16)tile[tx][nl];
  }
}

// ------------- 4) gate_up GEMM + fused SwiGLU + combine scaling ------------
// Block tile 128(M) x 64(N-i) x 32(K); 8 waves in 4x2, wave tile 32x32,
// dual accumulators (gate & up). Double-buffered async-to-LDS pipeline,
// statically unrolled x2. h'[t][e*960+i] = scale * silu(g) * u  (f16).
__global__ __launch_bounds__(256) void gemm_gate_up_swiglu(
    const _Float16* __restrict__ xf16, const _Float16* __restrict__ wgut,
    const float* __restrict__ comb, _Float16* __restrict__ hs) {
  __shared__ _Float16 As[2][128 * LDK];
  __shared__ _Float16 Bg[2][64 * LDK];
  __shared__ _Float16 Bu[2][64 * LDK];
  const int tid  = threadIdx.x;
  const int lane = tid & 31;
  const int w    = tid >> 5;
  const int wM   = w >> 1;                 // 0..3
  const int wN   = w & 1;                  // 0..1
  const int e    = blockIdx.z;
  const size_t tBase = (size_t)blockIdx.x * 128;
  const int    iBase = blockIdx.y * 64;
  const _Float16* aSrc = xf16 + tBase * DDIM;
  const _Float16* gSrc = wgut + ((size_t)e * 1920 + iBase) * DDIM;
  const _Float16* uSrc = wgut + ((size_t)e * 1920 + IPAD + iBase) * DDIM;
  const v8f vz = {0.f, 0.f, 0.f, 0.f, 0.f, 0.f, 0.f, 0.f};
  v8f accG[2][2], accU[2][2];
#pragma unroll
  for (int mf = 0; mf < 2; ++mf)
#pragma unroll
    for (int nf = 0; nf < 2; ++nf) { accG[mf][nf] = vz; accU[mf][nf] = vz; }

  const int r0 = tid >> 2,          q0 = (tid & 3) * 8;          // A chunk 0 / B chunk
  const int r1 = (tid + 256) >> 2,  q1 = (tid & 3) * 8;          // A chunk 1
  uint32_t lA[2], lA1[2], lBg[2], lBu[2];
#pragma unroll
  for (int b = 0; b < 2; ++b) {
    lA[b]  = lds_off(As[b]) + (uint32_t)(r0 * LDK + q0) * 2;
    lA1[b] = lds_off(As[b]) + (uint32_t)(r1 * LDK + q1) * 2;
    lBg[b] = lds_off(Bg[b]) + (uint32_t)(r0 * LDK + q0) * 2;
    lBu[b] = lds_off(Bu[b]) + (uint32_t)(r0 * LDK + q0) * 2;
  }
  // running global pointers (advance by 32 halves per K-step)
  const _Float16* pa0 = aSrc + (size_t)r0 * DDIM + q0;
  const _Float16* pa1 = aSrc + (size_t)r1 * DDIM + q1;
  const _Float16* pg  = gSrc + (size_t)r0 * DDIM + q0;
  const _Float16* pu  = uSrc + (size_t)r0 * DDIM + q0;

  auto stage = [&](int nb) {               // nb is a compile-time constant at call sites
    async_ld16(lA[nb],  pa0);
    async_ld16(lA1[nb], pa1);
    async_ld16(lBg[nb], pg);
    async_ld16(lBu[nb], pu);
    __builtin_prefetch(pg + 32, 0, 0);
    __builtin_prefetch(pu + 32, 0, 0);
    pa0 += 32; pa1 += 32; pg += 32; pu += 32;
  };
  auto compute = [&](int b) {
    const _Float16* aT = As[b];
    const _Float16* gT = Bg[b];
    const _Float16* uT = Bu[b];
    v16h aF[2], gF[2], uF[2];
#pragma unroll
    for (int mf = 0; mf < 2; ++mf) aF[mf] = frag_a(aT + (wM * 32 + mf * 16) * LDK, lane);
#pragma unroll
    for (int nf = 0; nf < 2; ++nf) {
      gF[nf] = frag_b(gT + (wN * 32 + nf * 16) * LDK, lane);
      uF[nf] = frag_b(uT + (wN * 32 + nf * 16) * LDK, lane);
    }
#pragma unroll
    for (int mf = 0; mf < 2; ++mf)
#pragma unroll
      for (int nf = 0; nf < 2; ++nf) {
        accG[mf][nf] = __builtin_amdgcn_wmma_f32_16x16x32_f16(
            false, aF[mf], false, gF[nf], (short)0, accG[mf][nf], false, false);
        accU[mf][nf] = __builtin_amdgcn_wmma_f32_16x16x32_f16(
            false, aF[mf], false, uF[nf], (short)0, accU[mf][nf], false, false);
      }
  };

  // software pipeline: 32 K-steps, unrolled x2, prologue + tail peeled
  stage(0);                                // step 0 -> buf0
  for (int it = 0; it < DDIM / 64 - 1; ++it) {
    stage(1); async_wait4(); __syncthreads(); compute(0); __syncthreads();
    stage(0); async_wait4(); __syncthreads(); compute(1); __syncthreads();
  }
  stage(1); async_wait4(); __syncthreads(); compute(0); __syncthreads();
  async_wait0();           __syncthreads(); compute(1);

  // epilogue: h = silu(g)*u*combine  (f16 store). C layout: M=v+8*(lane>=16), N=lane&15.
  const int colF   = lane & 15;
  const int rowOff = (lane & 16) ? 8 : 0;
#pragma unroll
  for (int mf = 0; mf < 2; ++mf) {
#pragma unroll
    for (int nf = 0; nf < 2; ++nf) {
      const int iCol = iBase + wN * 32 + nf * 16 + colF;
#pragma unroll
      for (int v = 0; v < 8; ++v) {
        const size_t t  = tBase + wM * 32 + mf * 16 + rowOff + v;
        const float  sc = comb[t * CSTRIDE + e];
        const float  g  = accG[mf][nf][v];
        const float  u  = accU[mf][nf][v];
        const float  sig = __builtin_amdgcn_rcpf(1.0f + __expf(-g));  // fast sigmoid
        const float  hv  = g * sig * u * sc;
        hs[t * KTOT + (size_t)e * IPAD + iCol] = (_Float16)hv;
      }
    }
  }
}

// ---------------- 5) down GEMM: out[4096,1024] = h'[4096,8640] @ W ---------
// Block tile 128(M) x 128(N) x 32(K); 8 waves in 4x2, wave tile 32x64
// (2 Mfrag x 4 Nfrag -> 8 WMMA per K-step per wave). Double-buffered async,
// statically unrolled x2.
__global__ __launch_bounds__(256) void gemm_down(
    const _Float16* __restrict__ hs, const _Float16* __restrict__ wdnt,
    float* __restrict__ out) {
  __shared__ _Float16 As[2][128 * LDK];
  __shared__ _Float16 Bs[2][128 * LDK];
  const int tid  = threadIdx.x;
  const int lane = tid & 31;
  const int w    = tid >> 5;
  const int wM   = w >> 1;                 // 0..3
  const int wN   = w & 1;                  // 0..1
  const size_t tBase = (size_t)blockIdx.x * 128;
  const int    nBase = blockIdx.y * 128;
  const _Float16* aSrc = hs   + tBase * KTOT;
  const _Float16* bSrc = wdnt + (size_t)nBase * KTOT;
  const v8f vz = {0.f, 0.f, 0.f, 0.f, 0.f, 0.f, 0.f, 0.f};
  v8f acc[2][4];
#pragma unroll
  for (int mf = 0; mf < 2; ++mf)
#pragma unroll
    for (int nf = 0; nf < 4; ++nf) acc[mf][nf] = vz;

  const int r0 = tid >> 2,         q0 = (tid & 3) * 8;
  const int r1 = (tid + 256) >> 2, q1 = (tid & 3) * 8;
  uint32_t lA0[2], lA1[2], lB0[2], lB1[2];
#pragma unroll
  for (int b = 0; b < 2; ++b) {
    lA0[b] = lds_off(As[b]) + (uint32_t)(r0 * LDK + q0) * 2;
    lA1[b] = lds_off(As[b]) + (uint32_t)(r1 * LDK + q1) * 2;
    lB0[b] = lds_off(Bs[b]) + (uint32_t)(r0 * LDK + q0) * 2;
    lB1[b] = lds_off(Bs[b]) + (uint32_t)(r1 * LDK + q1) * 2;
  }
  const _Float16* pa0 = aSrc + (size_t)r0 * KTOT + q0;
  const _Float16* pa1 = aSrc + (size_t)r1 * KTOT + q1;
  const _Float16* pb0 = bSrc + (size_t)r0 * KTOT + q0;
  const _Float16* pb1 = bSrc + (size_t)r1 * KTOT + q1;

  auto stage = [&](int nb) {
    async_ld16(lA0[nb], pa0);
    async_ld16(lA1[nb], pa1);
    async_ld16(lB0[nb], pb0);
    async_ld16(lB1[nb], pb1);
    __builtin_prefetch(pb0 + 32, 0, 0);
    __builtin_prefetch(pa1 + 32, 0, 0);
    pa0 += 32; pa1 += 32; pb0 += 32; pb1 += 32;
  };
  auto compute = [&](int b) {
    const _Float16* aT = As[b];
    const _Float16* bT = Bs[b];
    v16h aF[2], bF[4];
#pragma unroll
    for (int mf = 0; mf < 2; ++mf) aF[mf] = frag_a(aT + (wM * 32 + mf * 16) * LDK, lane);
#pragma unroll
    for (int nf = 0; nf < 4; ++nf) bF[nf] = frag_b(bT + (wN * 64 + nf * 16) * LDK, lane);
#pragma unroll
    for (int mf = 0; mf < 2; ++mf)
#pragma unroll
      for (int nf = 0; nf < 4; ++nf)
        acc[mf][nf] = __builtin_amdgcn_wmma_f32_16x16x32_f16(
            false, aF[mf], false, bF[nf], (short)0, acc[mf][nf], false, false);
  };

  // software pipeline: 270 K-steps, unrolled x2, prologue + tail peeled
  stage(0);                                // step 0 -> buf0
  for (int it = 0; it < KTOT / 64 - 1; ++it) {
    stage(1); async_wait4(); __syncthreads(); compute(0); __syncthreads();
    stage(0); async_wait4(); __syncthreads(); compute(1); __syncthreads();
  }
  stage(1); async_wait4(); __syncthreads(); compute(0); __syncthreads();
  async_wait0();           __syncthreads(); compute(1);

  const int colF   = lane & 15;
  const int rowOff = (lane & 16) ? 8 : 0;
#pragma unroll
  for (int mf = 0; mf < 2; ++mf)
#pragma unroll
    for (int nf = 0; nf < 4; ++nf) {
      const int n = nBase + wN * 64 + nf * 16 + colF;
#pragma unroll
      for (int v = 0; v < 8; ++v) {
        const size_t t = tBase + wM * 32 + mf * 16 + rowOff + v;
        out[t * DDIM + n] = acc[mf][nf][v];
      }
    }
}

// ------------------------------- launcher ----------------------------------
extern "C" void kernel_launch(void* const* d_in, const int* in_sizes, int n_in,
                              void* d_out, int out_size, void* d_ws, size_t ws_size,
                              hipStream_t stream) {
  const float* x   = (const float*)d_in[0];   // [2,2048,1024]
  const float* gw  = (const float*)d_in[1];   // [8,1024]
  const float* sgu = (const float*)d_in[2];   // [1024,1876]
  const float* sdn = (const float*)d_in[3];   // [938,1024]
  const float* egu = (const float*)d_in[4];   // [8,1024,1876]
  const float* edn = (const float*)d_in[5];   // [8,938,1024]
  float* out = (float*)d_out;

  char* ws = (char*)d_ws;
  _Float16* xf16 = (_Float16*)(ws);                    //  8,388,608 B
  float*    comb = (float*)   (ws + 8388608);          //    262,144 B
  _Float16* wgut = (_Float16*)(ws + 8650752);          // 35,389,440 B  [9][1920][1024]
  _Float16* wdnt = (_Float16*)(ws + 44040192);         // 17,694,720 B  [1024][8640]
  _Float16* hs   = (_Float16*)(ws + 61734912);         // 70,778,880 B  [4096][8640]

  router_convert<<<T_TOK, 128, 0, stream>>>(x, gw, xf16, comb);
  transpose_gu<<<dim3(60, 32, 9), dim3(32, 8), 0, stream>>>(sgu, egu, wgut);
  transpose_dn<<<dim3(30, 32, 9), dim3(32, 8), 0, stream>>>(sdn, edn, wdnt);
  gemm_gate_up_swiglu<<<dim3(32, 15, 9), 256, 0, stream>>>(xf16, wgut, comb, hs);
  gemm_down<<<dim3(32, 8), 256, 0, stream>>>(hs, wdnt, out);
}
